// MyDeepGCN_60885456388922
// MI455X (gfx1250) — matrix operand order repaired and checked by reading, not compiled, not run on hardware
//
#include <hip/hip_runtime.h>
#include <hip/hip_bf16.h>
#include <stdint.h>

#define NNODES 50000
#define NEDGE  400000
#define ETOT   (NNODES + NEDGE)   // edges + self loops = 450000
#define DCH    128
#define HCH    256                // H * C
#define NBLK   3

typedef __attribute__((ext_vector_type(16))) _Float16 v16h;
typedef __attribute__((ext_vector_type(8)))  float    v8f;

// ---------------- helpers ----------------

__device__ __forceinline__ float wave_sum32(float v) {
#pragma unroll
  for (int m = 16; m; m >>= 1) v += __shfl_xor(v, m, 32);
  return v;
}

// monotone float<->int encoding for atomicMax on floats
__device__ __forceinline__ int enc_ord(float f) {
  int i = __float_as_int(f);
  return i >= 0 ? i : (i ^ 0x7FFFFFFF);
}
__device__ __forceinline__ float dec_ord(int i) {
  return __int_as_float(i >= 0 ? i : (i ^ 0x7FFFFFFF));
}

// ---------------- init / convert ----------------

__global__ __launch_bounds__(256) void fill_f4_kernel(float4* __restrict__ p, int n4, float v) {
  int i = blockIdx.x * 256 + threadIdx.x;
  if (i < n4) p[i] = make_float4(v, v, v, v);
}

__global__ __launch_bounds__(256) void fill_i4_kernel(int4* __restrict__ p, int n4, int v) {
  int i = blockIdx.x * 256 + threadIdx.x;
  if (i < n4) p[i] = make_int4(v, v, v, v);
}

__global__ __launch_bounds__(256) void cvt_f16_kernel(const float* __restrict__ src,
                                                      _Float16* __restrict__ dst, int n) {
  int i = blockIdx.x * 256 + threadIdx.x;
  if (i < n) dst[i] = (_Float16)src[i];
}

// ---------------- LayerNorm + ReLU + f16 convert (wave per node) ----------------

__global__ __launch_bounds__(256) void ln_relu_kernel(const float* __restrict__ x,
                                                      const float* __restrict__ g,
                                                      const float* __restrict__ bt,
                                                      _Float16* __restrict__ xnh) {
  int wave = threadIdx.x >> 5, lane = threadIdx.x & 31;
  int node = blockIdx.x * 8 + wave;
  if (node >= NNODES) return;
  float4 v = *(const float4*)(x + (size_t)node * DCH + lane * 4);
  float mu = wave_sum32(v.x + v.y + v.z + v.w) * (1.0f / DCH);
  float dx = v.x - mu, dy = v.y - mu, dz = v.z - mu, dw = v.w - mu;
  float var = wave_sum32(dx * dx + dy * dy + dz * dz + dw * dw) * (1.0f / DCH);
  float rstd = rsqrtf(var + 1e-5f);
  float4 gv = *(const float4*)(g + lane * 4);
  float4 bv = *(const float4*)(bt + lane * 4);
  float y0 = dx * rstd * gv.x + bv.x;
  float y1 = dy * rstd * gv.y + bv.y;
  float y2 = dz * rstd * gv.z + bv.z;
  float y3 = dw * rstd * gv.w + bv.w;
  y0 = y0 > 0.f ? y0 : 0.f;
  y1 = y1 > 0.f ? y1 : 0.f;
  y2 = y2 > 0.f ? y2 : 0.f;
  y3 = y3 > 0.f ? y3 : 0.f;
  union { _Float16 h[4]; uint2 u2; } o;
  o.h[0] = (_Float16)y0; o.h[1] = (_Float16)y1;
  o.h[2] = (_Float16)y2; o.h[3] = (_Float16)y3;
  *(uint2*)(xnh + (size_t)node * DCH + lane * 4) = o.u2;
}

// ---------------- WMMA GEMM: H = XN(f16) @ W(f16), fp32 accum ----------------
// block = 256 threads = 8 waves; block tile = 128 rows x 64 cols.
// Whole K (=128) staged once: A [128][128] + B^T [64][128] in LDS with
// 136-half padded row stride -> conflict-free, 16B-aligned ds_load_b128
// fragment reads. One barrier, then 4 K-steps x 4 col-tiles of
// v_wmma_f32_16x16x32_f16 per wave with no further syncs.

#define ASTR 136   // padded LDS row stride (halves): 272B = 68 dwords -> 4-bank row skew
#define BSTR 136

__global__ __launch_bounds__(256) void gemm_wmma_kernel(const _Float16* __restrict__ A,   // [NNODES][128]
                                                        const _Float16* __restrict__ Bw,  // [128][256]
                                                        float* __restrict__ Hout) {       // [NNODES][256]
  __shared__ _Float16 Alds[128 * ASTR];  // [row][k]   ~34 KB
  __shared__ _Float16 Blds[64 * BSTR];   // [col][k]   ~17 KB (transposed W tile)

  const int tid = threadIdx.x;
  const int wave = tid >> 5, lane = tid & 31;
  const int row_base = blockIdx.x * 128;
  const int col_base = blockIdx.y * 64;

  // stage A: 128 rows x 128 k halves = 2048 uint4, 8 per thread (coalesced 16B)
#pragma unroll
  for (int li = 0; li < 8; ++li) {
    int idx = tid + li * 256;
    int r = idx >> 3, seg = idx & 7;       // 8 x uint4 per 128-half row
    int grow = row_base + r;
    uint4 val = make_uint4(0u, 0u, 0u, 0u);
    if (grow < NNODES) val = *(const uint4*)(A + (size_t)grow * DCH + seg * 8);
    *(uint4*)(Alds + r * ASTR + seg * 8) = val;
  }
  // stage B transposed: Bw rows k=0..127, cols col_base..col_base+63
  // 1024 vector reads of 8 halves, 4 per thread; scatter halves into [col][k]
#pragma unroll
  for (int li = 0; li < 4; ++li) {
    int idx = tid + li * 256;
    int r = idx >> 3, seg = idx & 7;       // r = k row, seg = col octet
    const _Float16* src = Bw + (size_t)r * HCH + col_base + seg * 8;
#pragma unroll
    for (int i = 0; i < 8; ++i) Blds[(seg * 8 + i) * BSTR + r] = src[i];
  }
  __syncthreads();

  v8f acc[4];
#pragma unroll
  for (int ct = 0; ct < 4; ++ct) acc[ct] = (v8f){};

  // fragment lane mapping, contiguous k = 16*(lane>>4) + i for A and B (16-bit A 16x32, B 32x16)
  const int arow = (wave << 4) + (lane & 15);
  const int akb = (lane >> 4) * 16;   // lanes 0-15: K 0..15; lanes 16-31: K 16..31
  const int bcol = lane & 15;
  const int bkb = (lane >> 4) * 16;   // lanes 0-15: K 0..15; lanes 16-31: K 16..31

#pragma unroll
  for (int kk = 0; kk < DCH; kk += 32) {
    v16h afrag;
#pragma unroll
    for (int i = 0; i < 8; ++i) {
      afrag[i]     = Alds[arow * ASTR + kk + akb + i];
      afrag[8 + i] = Alds[arow * ASTR + kk + 8 + akb + i];
    }
#pragma unroll
    for (int ct = 0; ct < 4; ++ct) {
      v16h bfrag;
#pragma unroll
      for (int i = 0; i < 16; ++i)
        bfrag[i] = Blds[(ct * 16 + bcol) * BSTR + kk + bkb + i];
      acc[ct] = __builtin_amdgcn_wmma_f32_16x16x32_f16(
          /*neg_a=*/false, afrag, /*neg_b=*/false, bfrag,
          /*c_mod=*/(short)0, acc[ct], /*reuse_a=*/false, /*reuse_b=*/false);
    }
  }

  // store C/D: VGPR i -> M = i + (lane>>4)*8, N = lane&15
  const int orow0 = row_base + (wave << 4) + ((lane >> 4) << 3);
  const int ocol0 = col_base + (lane & 15);
#pragma unroll
  for (int ct = 0; ct < 4; ++ct) {
#pragma unroll
    for (int i = 0; i < 8; ++i) {
      int r = orow0 + i;
      if (r < NNODES) Hout[(size_t)r * HCH + ocol0 + ct * 16] = acc[ct][i];
    }
  }
}

// ---------------- attention coefficients (wave per node) ----------------

__global__ __launch_bounds__(256) void alpha_kernel(const float* __restrict__ Hh,
                                                    const float* __restrict__ as,
                                                    const float* __restrict__ ad,
                                                    float* __restrict__ asrc,
                                                    float* __restrict__ adst) {
  int wave = threadIdx.x >> 5, lane = threadIdx.x & 31;
  int node = blockIdx.x * 8 + wave;
  if (node >= NNODES) return;
#pragma unroll
  for (int hd = 0; hd < 2; ++hd) {
    float4 hv = *(const float4*)(Hh + (size_t)node * HCH + hd * DCH + lane * 4);
    float4 sv = *(const float4*)(as + hd * DCH + lane * 4);
    float4 dv = *(const float4*)(ad + hd * DCH + lane * 4);
    float ps = wave_sum32(hv.x * sv.x + hv.y * sv.y + hv.z * sv.z + hv.w * sv.w);
    float pd = wave_sum32(hv.x * dv.x + hv.y * dv.y + hv.z * dv.z + hv.w * dv.w);
    if (lane == 0) { asrc[node * 2 + hd] = ps; adst[node * 2 + hd] = pd; }
  }
}

// ---------------- edge softmax: logits + segment max ----------------

__global__ __launch_bounds__(256) void edge_logits_max_kernel(const int* __restrict__ ei,
                                                              const float* __restrict__ asrc,
                                                              const float* __restrict__ adst,
                                                              float* __restrict__ elog,
                                                              int* __restrict__ mbuf) {
  int e = blockIdx.x * 256 + threadIdx.x;
  if (e >= ETOT) return;
  int s, d;
  if (e < NEDGE) { s = ei[e]; d = ei[NEDGE + e]; } else { s = e - NEDGE; d = s; }
#pragma unroll
  for (int hd = 0; hd < 2; ++hd) {
    float v = asrc[s * 2 + hd] + adst[d * 2 + hd];
    v = v > 0.f ? v : 0.2f * v;  // leaky relu
    elog[(size_t)e * 2 + hd] = v;
    atomicMax(&mbuf[d * 2 + hd], enc_ord(v));
  }
}

// ---------------- edge softmax: exp + segment sum ----------------

__global__ __launch_bounds__(256) void edge_exp_sum_kernel(const int* __restrict__ ei,
                                                           const float* __restrict__ elog,
                                                           const int* __restrict__ mbuf,
                                                           float* __restrict__ eexp,
                                                           float* __restrict__ sbuf) {
  int e = blockIdx.x * 256 + threadIdx.x;
  if (e >= ETOT) return;
  int d = (e < NEDGE) ? ei[NEDGE + e] : (e - NEDGE);
#pragma unroll
  for (int hd = 0; hd < 2; ++hd) {
    float ex = expf(elog[(size_t)e * 2 + hd] - dec_ord(mbuf[d * 2 + hd]));
    eexp[(size_t)e * 2 + hd] = ex;
    atomicAdd(&sbuf[d * 2 + hd], ex);
  }
}

// ---------------- weighted message scatter (wave per edge, 256 ch) ----------------

__global__ __launch_bounds__(256) void scatter_msgs_kernel(const int* __restrict__ ei,
                                                           const float* __restrict__ Hh,
                                                           const float* __restrict__ eexp,
                                                           const float* __restrict__ sbuf,
                                                           float* __restrict__ acc) {
  int wave = threadIdx.x >> 5, lane = threadIdx.x & 31;
  int e = blockIdx.x * 8 + wave;
  if (e >= ETOT) return;
  int s, d;
  if (e < NEDGE) { s = ei[e]; d = ei[NEDGE + e]; } else { s = e - NEDGE; d = s; }
  int hd = lane >> 4;  // lanes 0-15 -> channels 0..127 (head 0), 16-31 -> head 1
  float al = eexp[(size_t)e * 2 + hd] / (sbuf[d * 2 + hd] + 1e-16f);
  const float4* hp = (const float4*)(Hh + (size_t)s * HCH + lane * 8);
  float4 v0 = hp[0], v1 = hp[1];
  float* ap = acc + (size_t)d * HCH + lane * 8;
  atomicAdd(ap + 0, v0.x * al);
  atomicAdd(ap + 1, v0.y * al);
  atomicAdd(ap + 2, v0.z * al);
  atomicAdd(ap + 3, v0.w * al);
  atomicAdd(ap + 4, v1.x * al);
  atomicAdd(ap + 5, v1.y * al);
  atomicAdd(ap + 6, v1.z * al);
  atomicAdd(ap + 7, v1.w * al);
}

// ---------------- head mean + bias + residual ----------------

__global__ __launch_bounds__(256) void finalize_kernel(const float* __restrict__ acc,
                                                       const float* __restrict__ bias,
                                                       const float* __restrict__ xin,
                                                       float* __restrict__ xout) {
  int i = blockIdx.x * 256 + threadIdx.x;
  if (i >= NNODES * DCH) return;
  int n = i >> 7, c = i & 127;
  xout[i] = 0.5f * (acc[(size_t)n * HCH + c] + acc[(size_t)n * HCH + DCH + c]) + bias[c] + xin[i];
}

// ---------------- host launcher ----------------

extern "C" void kernel_launch(void* const* d_in, const int* in_sizes, int n_in,
                              void* d_out, int out_size, void* d_ws, size_t ws_size,
                              hipStream_t stream) {
  (void)in_sizes; (void)n_in; (void)out_size; (void)ws_size;
  const float* x     = (const float*)d_in[0];
  const int*   ei    = (const int*)d_in[1];
  const float* W     = (const float*)d_in[2];
  const float* a_s   = (const float*)d_in[3];
  const float* a_d   = (const float*)d_in[4];
  const float* bias  = (const float*)d_in[5];
  const float* gamma = (const float*)d_in[6];
  const float* beta  = (const float*)d_in[7];
  float* out = (float*)d_out;

  char* wsp = (char*)d_ws;
  size_t off = 0;
  auto walloc = [&](size_t bytes) -> char* {
    char* p = wsp + off;
    off += (bytes + 255) & ~(size_t)255;
    return p;
  };

  _Float16* Wh   = (_Float16*)walloc((size_t)NBLK * DCH * HCH * 2);
  _Float16* xnh  = (_Float16*)walloc((size_t)NNODES * DCH * 2);
  float*    hbuf = (float*)walloc((size_t)NNODES * HCH * 4);
  float*    acc  = (float*)walloc((size_t)NNODES * HCH * 4);
  float*    asrc = (float*)walloc((size_t)NNODES * 2 * 4);
  float*    adst = (float*)walloc((size_t)NNODES * 2 * 4);
  int*      mbuf = (int*)walloc((size_t)NNODES * 2 * 4);
  float*    sbuf = (float*)walloc((size_t)NNODES * 2 * 4);
  float*    elog = (float*)walloc((size_t)ETOT * 2 * 4);
  float*    eexp = (float*)walloc((size_t)ETOT * 2 * 4);
  float*    xa   = (float*)walloc((size_t)NNODES * DCH * 4);
  float*    xb   = (float*)walloc((size_t)NNODES * DCH * 4);

  // per-call fp16 conversion of all 3 weight matrices
  const int wtot = NBLK * DCH * HCH;
  cvt_f16_kernel<<<(wtot + 255) / 256, 256, 0, stream>>>(W, Wh, wtot);

  const int ENC_NEG_INF = 0x807FFFFF;  // enc_ord(-inf)

  const float* xin = x;
  for (int b = 0; b < NBLK; ++b) {
    float* xout = (b == 0) ? xa : (b == 1) ? xb : out;

    fill_f4_kernel<<<(NNODES * HCH / 4 + 255) / 256, 256, 0, stream>>>((float4*)acc, NNODES * HCH / 4, 0.f);
    fill_f4_kernel<<<(NNODES * 2 / 4 + 255) / 256, 256, 0, stream>>>((float4*)sbuf, NNODES * 2 / 4, 0.f);
    fill_i4_kernel<<<(NNODES * 2 / 4 + 255) / 256, 256, 0, stream>>>((int4*)mbuf, NNODES * 2 / 4, ENC_NEG_INF);

    ln_relu_kernel<<<(NNODES + 7) / 8, 256, 0, stream>>>(xin, gamma + b * DCH, beta + b * DCH, xnh);

    dim3 gemm_grid((NNODES + 127) / 128, HCH / 64);
    gemm_wmma_kernel<<<gemm_grid, 256, 0, stream>>>(xnh, Wh + (size_t)b * DCH * HCH, hbuf);

    alpha_kernel<<<(NNODES + 7) / 8, 256, 0, stream>>>(hbuf, a_s + b * HCH, a_d + b * HCH, asrc, adst);

    edge_logits_max_kernel<<<(ETOT + 255) / 256, 256, 0, stream>>>(ei, asrc, adst, elog, mbuf);
    edge_exp_sum_kernel<<<(ETOT + 255) / 256, 256, 0, stream>>>(ei, elog, mbuf, eexp, sbuf);
    scatter_msgs_kernel<<<(ETOT + 7) / 8, 256, 0, stream>>>(ei, hbuf, eexp, sbuf, acc);

    finalize_kernel<<<(NNODES * DCH + 255) / 256, 256, 0, stream>>>(acc, bias + b * DCH, xin, xout);

    xin = xout;
  }
}
